// ReadHead_2783138808152
// MI455X (gfx1250) — matrix operand order repaired and checked by reading, not compiled, not run on hardware
//
#include <hip/hip_runtime.h>
#include <hip/hip_bf16.h>

typedef __attribute__((ext_vector_type(16))) __bf16 v16bf;
typedef __attribute__((ext_vector_type(8)))  __bf16 v8bf;
typedef __attribute__((ext_vector_type(4)))  __bf16 v4bf;
typedef __attribute__((ext_vector_type(2)))  __bf16 v2bf;
typedef __attribute__((ext_vector_type(8)))  float  v8f;

constexpr int Bdim = 8192;   // batch
constexpr int Cdim = 1024;   // controller width (K of GEMM)
constexpr int Mdim = 4096;   // memory width (N of GEMM)

constexpr int ROWS   = 128;  // rows per workgroup
constexpr int NCHUNK = 128;  // logit columns per chunk (2 waves x 64)
constexpr int KSTAGE = 64;   // K depth staged in LDS per buffer
constexpr int NK     = Cdim / KSTAGE;   // 16 stages
constexpr int APAD   = 72;   // padded k-stride in bf16 elems (144B = 36 banks,
                             // gcd(36,64)=4 -> conflict-free b128 tile reads)

// ---------------------------------------------------------------------------
// Kernel 1: fused  logits = A@W + b  ->  online softmax stats  ->  cosine vs
// memory. Never materializes logits. Double-buffered LDS staging: one barrier
// per K-stage; next tile's global loads overlap current tile's WMMAs.
// ---------------------------------------------------------------------------
__global__ __launch_bounds__(256)
void fused_keysim_kernel(const float* __restrict__ mem,   // [B, M]
                         const float* __restrict__ A,     // [B, C] controller_output
                         const float* __restrict__ W,     // [C, M]
                         const float* __restrict__ bias,  // [M]
                         float* __restrict__ sims,        // [B]
                         float* __restrict__ rowsum)      // [B]
{
    __shared__ __attribute__((aligned(16))) __bf16 Abf[2][ROWS][APAD];   // A tiles
    __shared__ __attribute__((aligned(16))) __bf16 Wt[2][NCHUNK][APAD];  // W^T tiles
    __shared__ float wvMax[2][ROWS], wvS2[2][ROWS], wvD[2][ROWS];
    __shared__ float wvMS[2][ROWS], wvM2[2][ROWS];
    __shared__ float rsM[ROWS], rsS2[ROWS], rsD[ROWS], rsMS[ROWS], rsM2[ROWS];

    const int tid  = threadIdx.x;
    const int lane = tid & 31;
    const int wave = tid >> 5;         // 0..7
    const int wr   = wave >> 1;        // 0..3 row quadrant
    const int wc   = wave & 1;         // 0..1 column half
    const int rb   = wr * 32;          // wave row base within 128-row block
    const int cb   = wc * 64;          // wave col base within 128-col chunk
    const int hl   = (lane < 16) ? 0 : 1;
    const int l16  = lane & 15;
    const int gr0  = blockIdx.x * ROWS;

    // ---- staging helpers (fp32 global -> bf16 LDS, packed stores) ----
    auto stageA = [&](int buf, int kk) {
        #pragma unroll
        for (int p = 0; p < 8; ++p) {
            const int rr = p * 16 + (tid >> 4);     // 16 rows per pass
            const int kq = (tid & 15) * 4;          // 4 floats per thread
            float4 f = *(const float4*)(A + (size_t)(gr0 + rr) * Cdim + kk + kq);
            v4bf hv = { (__bf16)f.x, (__bf16)f.y, (__bf16)f.z, (__bf16)f.w };
            *(v4bf*)&Abf[buf][rr][kq] = hv;         // one ds_store_b64
        }
    };
    auto stageW = [&](int buf, int kk, int nn) {
        #pragma unroll
        for (int it = 0; it < 16; ++it) {
            const int kl = ((tid >> 7) + it * 2) * 2;  // even k, 0..62
            const int nl = tid & 127;                  // coalesced along n
            const float f0 = W[(size_t)(kk + kl)     * Mdim + nn + nl];
            const float f1 = W[(size_t)(kk + kl + 1) * Mdim + nn + nl];
            v2bf hv = { (__bf16)f0, (__bf16)f1 };
            *(v2bf*)&Wt[buf][nl][kl] = hv;             // one ds_store_b32
        }
    };

    if (tid < ROWS) {
        rsM[tid] = -__builtin_inff();
        rsS2[tid] = 0.f; rsD[tid] = 0.f; rsMS[tid] = 0.f; rsM2[tid] = 0.f;
    }
    __syncthreads();

    for (int nn = 0; nn < Mdim; nn += NCHUNK) {
        v8f acc[2][4] = {};   // fp32 accumulators: 2 row-tiles x 4 col-tiles

        stageA(0, 0);
        stageW(0, 0, nn);
        __syncthreads();

        for (int kt = 0; kt < NK; ++kt) {
            const int cur = kt & 1;
            // prefetch next stage into the other buffer (overlaps WMMAs below)
            if (kt + 1 < NK) {
                stageA(cur ^ 1, (kt + 1) * KSTAGE);
                stageW(cur ^ 1, (kt + 1) * KSTAGE, nn);
            }

            // ---- two WMMA k-steps from the current staged tile ----
            #pragma unroll
            for (int s = 0; s < 2; ++s) {
                const int ks = s * 32;
                v16bf afrag[2], bfrag[4];
                const int aoff = ks + (hl ? 8 : 0);   // A lanes 0-15: K{0..7,16..23}
                #pragma unroll
                for (int i = 0; i < 2; ++i) {
                    const int row = rb + i * 16 + l16;
                    ((v8bf*)&afrag[i])[0] = *(const v8bf*)&Abf[cur][row][aoff];
                    ((v8bf*)&afrag[i])[1] = *(const v8bf*)&Abf[cur][row][16 + aoff];
                }
                const int boff = ks + (hl ? 8 : 0);   // B lanes 0-15: K{0..7,16..23}
                #pragma unroll
                for (int j = 0; j < 4; ++j) {
                    const int col = cb + j * 16 + l16;
                    ((v8bf*)&bfrag[j])[0] = *(const v8bf*)&Wt[cur][col][boff];
                    ((v8bf*)&bfrag[j])[1] = *(const v8bf*)&Wt[cur][col][boff + 16];
                }
                #pragma unroll
                for (int i = 0; i < 2; ++i)
                    #pragma unroll
                    for (int j = 0; j < 4; ++j)
                        acc[i][j] = __builtin_amdgcn_wmma_f32_16x16x32_bf16(
                            false, afrag[i], false, bfrag[j],
                            (short)0, acc[i][j], false, false);
            }
            __syncthreads();   // next buffer staged AND current reads done
        }

        // ---- add bias (column = cb + j*16 + l16 for every accum element) ----
        float bj[4];
        #pragma unroll
        for (int j = 0; j < 4; ++j) bj[j] = bias[nn + cb + j * 16 + l16];
        #pragma unroll
        for (int i = 0; i < 2; ++i)
            #pragma unroll
            for (int j = 0; j < 4; ++j) {
                const float bb = bj[j];
                #pragma unroll
                for (int g = 0; g < 8; ++g) acc[i][j][g] += bb;
            }

        // ---- per-wave row max (row r = rb + i*16 + g + hl*8 lives entirely
        //      on one 16-lane half; shuffle-reduce across that half) ----
        #pragma unroll
        for (int i = 0; i < 2; ++i)
            #pragma unroll
            for (int g = 0; g < 8; ++g) {
                float pm = fmaxf(fmaxf(acc[i][0][g], acc[i][1][g]),
                                 fmaxf(acc[i][2][g], acc[i][3][g]));
                pm = fmaxf(pm, __shfl_xor(pm, 1));
                pm = fmaxf(pm, __shfl_xor(pm, 2));
                pm = fmaxf(pm, __shfl_xor(pm, 4));
                pm = fmaxf(pm, __shfl_xor(pm, 8));
                if (l16 == 0) wvMax[wc][rb + i * 16 + g + hl * 8] = pm;
            }
        __syncthreads();

        // ---- chunk partials: exp terms + streamed memory dot/sums ----
        #pragma unroll
        for (int i = 0; i < 2; ++i)
            #pragma unroll
            for (int g = 0; g < 8; ++g) {
                const int rloc = rb + i * 16 + g + hl * 8;
                const float cmax = fmaxf(wvMax[0][rloc], wvMax[1][rloc]);
                const float mnew = fmaxf(rsM[rloc], cmax);
                const float* mrow = mem + (size_t)(gr0 + rloc) * Mdim + nn;
                float s2p = 0.f, dp = 0.f, msp = 0.f, m2p = 0.f;
                #pragma unroll
                for (int j = 0; j < 4; ++j) {
                    const float e  = __expf(acc[i][j][g] - mnew);
                    const float mv = mrow[cb + j * 16 + l16];
                    s2p += e * e;      // e^{2(l-m)}
                    dp  += e * mv;     // e^{l-m} * mem
                    msp += mv;         // rowsum(mem)
                    m2p += mv * mv;    // ||mem||^2
                }
                #pragma unroll
                for (int s = 1; s <= 8; s <<= 1) {
                    s2p += __shfl_xor(s2p, s);
                    dp  += __shfl_xor(dp,  s);
                    msp += __shfl_xor(msp, s);
                    m2p += __shfl_xor(m2p, s);
                }
                if (l16 == 0) {
                    wvS2[wc][rloc] = s2p; wvD[wc][rloc] = dp;
                    wvMS[wc][rloc] = msp; wvM2[wc][rloc] = m2p;
                }
            }
        __syncthreads();

        // ---- online rescale of running stats (one thread per row) ----
        if (tid < ROWS) {
            const int r = tid;
            const float mold = rsM[r];
            const float mnew = fmaxf(mold, fmaxf(wvMax[0][r], wvMax[1][r]));
            const float al   = __expf(mold - mnew);   // 0 on first chunk
            rsS2[r] = rsS2[r] * al * al + wvS2[0][r] + wvS2[1][r];
            rsD[r]  = rsD[r]  * al      + wvD[0][r]  + wvD[1][r];
            rsMS[r] += wvMS[0][r] + wvMS[1][r];
            rsM2[r] += wvM2[0][r] + wvM2[1][r];
            rsM[r] = mnew;
        }
        __syncthreads();
    }

    // sims = -<key_hat, mem_hat>; softmax denominator cancels inside l2-norm.
    if (tid < ROWS) {
        const float s2 = fmaxf(rsS2[tid], 1e-12f);
        const float m2 = fmaxf(rsM2[tid], 1e-12f);
        sims[gr0 + tid]   = -rsD[tid] * __frsqrt_rn(s2) * __frsqrt_rn(m2);
        rowsum[gr0 + tid] = rsMS[tid];
    }
}

// ---------------------------------------------------------------------------
// Kernel 2: softmax over the batch dimension + final scale by rowsum(mem).
// ---------------------------------------------------------------------------
__global__ __launch_bounds__(1024)
void softmax_read_kernel(const float* __restrict__ sims,
                         const float* __restrict__ rowsum,
                         float* __restrict__ out)
{
    __shared__ float red[32];
    const int tid = threadIdx.x, lane = tid & 31, wv = tid >> 5;

    float m = -__builtin_inff();
    for (int i = tid; i < Bdim; i += 1024) m = fmaxf(m, sims[i]);
    #pragma unroll
    for (int s = 16; s >= 1; s >>= 1) m = fmaxf(m, __shfl_xor(m, s));
    if (lane == 0) red[wv] = m;
    __syncthreads();
    float gm = red[0];
    #pragma unroll
    for (int w = 1; w < 32; ++w) gm = fmaxf(gm, red[w]);
    __syncthreads();

    float z = 0.f;
    for (int i = tid; i < Bdim; i += 1024) z += __expf(sims[i] - gm);
    #pragma unroll
    for (int s = 16; s >= 1; s >>= 1) z += __shfl_xor(z, s);
    if (lane == 0) red[wv] = z;
    __syncthreads();
    float gz = 0.f;
    #pragma unroll
    for (int w = 0; w < 32; ++w) gz += red[w];
    const float inv = 1.0f / gz;

    for (int i = tid; i < Bdim; i += 1024)
        out[i] = __expf(sims[i] - gm) * inv * rowsum[i];
}

// ---------------------------------------------------------------------------
extern "C" void kernel_launch(void* const* d_in, const int* in_sizes, int n_in,
                              void* d_out, int out_size, void* d_ws, size_t ws_size,
                              hipStream_t stream) {
    (void)in_sizes; (void)n_in; (void)out_size; (void)ws_size;
    const float* mem  = (const float*)d_in[0];   // memory            [B, M]
    const float* ctrl = (const float*)d_in[1];   // controller_output [B, C]
    const float* W    = (const float*)d_in[2];   // W                 [C, M]
    const float* bias = (const float*)d_in[3];   // b                 [M]
    float* out    = (float*)d_out;               // read_vector       [B]
    float* sims   = (float*)d_ws;                // [B]
    float* rowsum = sims + Bdim;                 // [B]

    fused_keysim_kernel<<<Bdim / ROWS, 256, 0, stream>>>(mem, ctrl, W, bias, sims, rowsum);
    softmax_read_kernel<<<1, 1024, 0, stream>>>(sims, rowsum, out);
}
